// Encoder_16063177687568
// MI455X (gfx1250) — matrix operand order repaired and checked by reading, not compiled, not run on hardware
//
#include <hip/hip_runtime.h>

// GNN encoder, 3 layers:  y = segment_sum(sig(z@Wf+bf)*softplus(z@Ws+bs), dst) + x@Wu + bu
// Edge GEMMs on v_wmma_f32_16x16x32_bf16 (f32 accumulate). Weights are staged in LDS
// (transposed + K-padded) once per block and reused across ~20 edge tiles, so the
// WMMA inner loop reads only LDS.

typedef __attribute__((ext_vector_type(16))) __bf16       v16bf;
typedef __attribute__((ext_vector_type(8)))  float        v8f;
typedef __attribute__((ext_vector_type(8)))  unsigned int v8u;

#define DEV __device__ __forceinline__

DEV unsigned short f2bf(float f) {               // round-to-nearest-even fp32 -> bf16
  unsigned int u = __builtin_bit_cast(unsigned int, f);
  u += 0x7FFFu + ((u >> 16) & 1u);
  return (unsigned short)(u >> 16);
}

__global__ void k_f32_to_bf16(const float* __restrict__ in,
                              unsigned short* __restrict__ out, int n) {
  int i = blockIdx.x * blockDim.x + threadIdx.x;
  if (i < n) out[i] = f2bf(in[i]);
}

// W [zin, dout] row-major  ->  Wt [dout, kpad] (transposed, K zero-padded to mult of 32)
__global__ void k_weight_bf16_T(const float* __restrict__ W,
                                unsigned short* __restrict__ Wt,
                                int zin, int dout, int kpad) {
  int i = blockIdx.x * blockDim.x + threadIdx.x;
  if (i >= dout * kpad) return;
  int n = i / kpad, k = i - n * kpad;
  Wt[i] = (k < zin) ? f2bf(W[k * dout + n]) : (unsigned short)0;
}

// y[n,o] = bu[o] + sum_k x[n,k] * Wu[k,o]   (tiny: ~1.3 GFLOP total, fp32)
template <int DIN, int DOUT>
__global__ void k_node_update(const float* __restrict__ x, const float* __restrict__ Wu,
                              const float* __restrict__ bu, float* __restrict__ y, int Nn) {
  int n = blockIdx.x;
  if (n >= Nn) return;
  int o = threadIdx.x;                           // blockDim == DOUT
  float acc = bu[o];
  const float* xr = x + (size_t)n * DIN;
#pragma unroll 4
  for (int k = 0; k < DIN; ++k) acc = fmaf(xr[k], Wu[k * DOUT + o], acc);
  y[(size_t)n * DOUT + o] = acc;
}

// Block = 512 threads (16 waves). blockIdx.x selects a 64-column slice of W
// (held in LDS for the whole block), blockIdx.y selects a chunk of edge tiles.
// Per iteration: 64 edges; wave w does m-tile (w&3), n-tile (w>>2).
template <int DIN, int DOUT>
__global__ __launch_bounds__(512) void k_edge_wmma(
    const unsigned short* __restrict__ xb,   // [N, DIN]  bf16 node features
    const unsigned short* __restrict__ eab,  // [E, 16]   bf16 edge attrs
    const int*            __restrict__ eidx, // [2, E]    src row then dst row
    const unsigned short* __restrict__ WfT,  // [DOUT, KPAD] bf16 (transposed, padded)
    const float*          __restrict__ bfv,  // [DOUT]
    const unsigned short* __restrict__ WsT,  // [DOUT, KPAD]
    const float*          __restrict__ bsv,  // [DOUT]
    float*                __restrict__ y,    // [N, DOUT]  accumulated via atomics
    int E_) {
  constexpr int ZIN  = 2 * DIN + 16;
  constexpr int KPAD = ((ZIN + 31) / 32) * 32;
  constexpr int COLS = 64;                   // column slice per block
  constexpr int KB   = KPAD / 32;

  __shared__ int            sdst[64];
  __shared__ unsigned short zs[64 * KPAD];          // 64-edge z tile
  __shared__ unsigned short wlds[2 * COLS * KPAD];  // [Wf | Ws] column slice

  const int t    = threadIdx.x;
  const int lane = t & 31;
  const int wave = t >> 5;
  const int m    = lane & 15;
  const int hi   = (lane >> 4) & 1;
  const int mt   = wave & 3;                 // m-tile 0..3
  const int ntl  = wave >> 2;                // n-tile 0..3
  const int n0   = blockIdx.x * COLS;

  // ---- stage W column slice into LDS once (contiguous in transposed layout) ----
  {
    const uint4* gF = (const uint4*)(WfT + (size_t)n0 * KPAD);
    const uint4* gS = (const uint4*)(WsT + (size_t)n0 * KPAD);
    uint4* wl = (uint4*)wlds;
    const int n16 = COLS * KPAD / 8;         // uint4 count per matrix
    for (int i = t; i < n16; i += 512) wl[i] = gF[i];
    for (int i = t; i < n16; i += 512) wl[n16 + i] = gS[i];
  }

  const int n   = n0 + ntl * 16 + m;         // this lane's output column
  const float bFv = bfv[n];
  const float bSv = bsv[n];

  const unsigned int* zs32 = (const unsigned int*)zs;
  const unsigned int* wf32 = (const unsigned int*)(wlds + (size_t)(ntl * 16 + m) * KPAD);
  const unsigned int* ws32 = (const unsigned int*)(wlds + (size_t)(COLS + ntl * 16 + m) * KPAD);

  const int tiles = (E_ + 63) >> 6;
  const int per   = (tiles + gridDim.y - 1) / gridDim.y;
  const int t0    = blockIdx.y * per;
  const int t1    = (t0 + per < tiles) ? (t0 + per) : tiles;

  const v8f zero = {};

  for (int tile = t0; tile < t1; ++tile) {
    const int e0 = tile << 6;
    __syncthreads();                         // zs / wlds(first iter) safe to (re)use

    // ---- gather: z = [x[dst] | x[src] | edge_attr] for 64 edges, uint-wide ----
    {
      const int r = t >> 3;                  // 64 rows, 8 threads per row
      const int e = e0 + r;
      int sidx = 0, didx = -1;
      if (e < E_) { sidx = eidx[e]; didx = eidx[E_ + e]; }
      if ((t & 7) == 0) sdst[r] = didx;
      const unsigned int* xd = (const unsigned int*)(xb + (size_t)(didx < 0 ? 0 : didx) * DIN);
      const unsigned int* xs = (const unsigned int*)(xb + (size_t)sidx * DIN);
      const unsigned int* ep = (const unsigned int*)(eab + (size_t)e * 16);
      unsigned int* zr = (unsigned int*)zs + (size_t)r * (KPAD / 2);
      for (int k2 = (t & 7); k2 < KPAD / 2; k2 += 8) {
        unsigned int v = 0;
        if (e < E_) {
          if (k2 < DIN / 2)        v = xd[k2];             // x[dst]
          else if (k2 < DIN)       v = xs[k2 - DIN / 2];   // x[src]
          else if (k2 < ZIN / 2)   v = ep[k2 - DIN];       // edge_attr
        }
        zr[k2] = v;
      }
    }
    __syncthreads();

    // ---- K loop: all operands in LDS ----
    v8f accF = zero, accS = zero;
    const int arow = (mt * 16 + m) * KPAD;
#pragma unroll 4
    for (int kb = 0; kb < KB; ++kb) {
      // A fragment (16x32 bf16): lanes 0-15 -> K 0..15, lanes 16-31 -> K 16..31
      // vgpr j : K=base+2j
      v8u araw;
      const int abase = (arow + kb * 32 + hi * 16) >> 1;
#pragma unroll
      for (int j = 0; j < 8; ++j) {
        const int k2 = j;
        araw[j] = zs32[abase + k2];
      }
      const v16bf a = __builtin_bit_cast(v16bf, araw);

      // B fragments (32x16 bf16): lanes 0-15 -> K 0..15, 16-31 -> K 16..31
      const int bbase = (kb * 32 + hi * 16) >> 1;
      v8u fraw, sraw;
#pragma unroll
      for (int j = 0; j < 8; ++j) { fraw[j] = wf32[bbase + j]; sraw[j] = ws32[bbase + j]; }
      const v16bf bF = __builtin_bit_cast(v16bf, fraw);
      const v16bf bS = __builtin_bit_cast(v16bf, sraw);

      accF = __builtin_amdgcn_wmma_f32_16x16x32_bf16(false, a, false, bF, (short)0, accF,
                                                     false, false);
      accS = __builtin_amdgcn_wmma_f32_16x16x32_bf16(false, a, false, bS, (short)0, accS,
                                                     false, false);
    }

    // ---- epilogue: bias, sigmoid*softplus, scatter-add (segment_sum) ----
#pragma unroll
    for (int r = 0; r < 8; ++r) {
      const int mrow = mt * 16 + r + hi * 8;  // C layout: M = r + 8*(lane>=16)
      const int d = sdst[mrow];
      if (d < 0) continue;
      const float f = accF[r] + bFv;
      const float s = accS[r] + bSv;
      const float sig = 1.0f / (1.0f + expf(-f));
      const float sp  = (s > 20.0f) ? s : log1pf(expf(s));
      atomicAdd(&y[(size_t)d * DOUT + n], sig * sp);
    }
  }
}

extern "C" void kernel_launch(void* const* d_in, const int* in_sizes, int n_in,
                              void* d_out, int out_size, void* d_ws, size_t ws_size,
                              hipStream_t stream) {
  const float* x    = (const float*)d_in[0];
  const int*   eidx = (const int*)d_in[1];     // [2,E]: row0=src, row1=dst
  const float* ea   = (const float*)d_in[2];
  const float* Wf0 = (const float*)d_in[4];  const float* bf0 = (const float*)d_in[5];
  const float* Ws0 = (const float*)d_in[6];  const float* bs0 = (const float*)d_in[7];
  const float* Wu0 = (const float*)d_in[8];  const float* bu0 = (const float*)d_in[9];
  const float* Wf1 = (const float*)d_in[10]; const float* bf1 = (const float*)d_in[11];
  const float* Ws1 = (const float*)d_in[12]; const float* bs1 = (const float*)d_in[13];
  const float* Wu1 = (const float*)d_in[14]; const float* bu1 = (const float*)d_in[15];
  const float* Wf2 = (const float*)d_in[16]; const float* bf2 = (const float*)d_in[17];
  const float* Ws2 = (const float*)d_in[18]; const float* bs2 = (const float*)d_in[19];
  const float* Wu2 = (const float*)d_in[20]; const float* bu2 = (const float*)d_in[21];
  float* out = (float*)d_out;

  const int N = in_sizes[0] / 64;     // 10000
  const int E = in_sizes[1] / 2;      // 160000

  // workspace carve-up (~31 MB total)
  char* ws = (char*)d_ws;
  auto up = [](size_t v) { return (v + 255) & ~(size_t)255; };
  unsigned short* xb  = (unsigned short*)ws; ws += up((size_t)N * 256 * 2);
  unsigned short* eab = (unsigned short*)ws; ws += up((size_t)E * 16 * 2);
  unsigned short* WfT = (unsigned short*)ws; ws += up((size_t)256 * 544 * 2);
  unsigned short* WsT = (unsigned short*)ws; ws += up((size_t)256 * 544 * 2);
  float* buf0 = (float*)ws;                  ws += up((size_t)N * 256 * 4);
  float* buf1 = (float*)ws;                  ws += up((size_t)N * 256 * 4);
  (void)ws_size; (void)n_in; (void)out_size;

  const int CT = 256;

  // edge attrs -> bf16 once
  k_f32_to_bf16<<<(E * 16 + CT - 1) / CT, CT, 0, stream>>>(ea, eab, E * 16);

  // ---------------- layer 0: din=64, dout=256, zin=144 (kpad=160) ----------------
  k_f32_to_bf16<<<(N * 64 + CT - 1) / CT, CT, 0, stream>>>(x, xb, N * 64);
  k_weight_bf16_T<<<(256 * 160 + CT - 1) / CT, CT, 0, stream>>>(Wf0, WfT, 144, 256, 160);
  k_weight_bf16_T<<<(256 * 160 + CT - 1) / CT, CT, 0, stream>>>(Ws0, WsT, 144, 256, 160);
  k_node_update<64, 256><<<N, 256, 0, stream>>>(x, Wu0, bu0, buf0, N);
  k_edge_wmma<64, 256><<<dim3(4, 128), 512, 0, stream>>>(xb, eab, eidx, WfT, bf0, WsT, bs0,
                                                         buf0, E);

  // ---------------- layer 1: din=256, dout=256, zin=528 (kpad=544) ----------------
  k_f32_to_bf16<<<(N * 256 + CT - 1) / CT, CT, 0, stream>>>(buf0, xb, N * 256);
  k_weight_bf16_T<<<(256 * 544 + CT - 1) / CT, CT, 0, stream>>>(Wf1, WfT, 528, 256, 544);
  k_weight_bf16_T<<<(256 * 544 + CT - 1) / CT, CT, 0, stream>>>(Ws1, WsT, 528, 256, 544);
  k_node_update<256, 256><<<N, 256, 0, stream>>>(buf0, Wu1, bu1, buf1, N);
  k_edge_wmma<256, 256><<<dim3(4, 128), 512, 0, stream>>>(xb, eab, eidx, WfT, bf1, WsT, bs1,
                                                          buf1, E);

  // ---------------- layer 2: din=256, dout=128, zin=528 (kpad=544) ----------------
  k_f32_to_bf16<<<(N * 256 + CT - 1) / CT, CT, 0, stream>>>(buf1, xb, N * 256);
  k_weight_bf16_T<<<(128 * 544 + CT - 1) / CT, CT, 0, stream>>>(Wf2, WfT, 528, 128, 544);
  k_weight_bf16_T<<<(128 * 544 + CT - 1) / CT, CT, 0, stream>>>(Ws2, WsT, 528, 128, 544);
  k_node_update<256, 128><<<N, 128, 0, stream>>>(buf1, Wu2, bu2, out, N);
  k_edge_wmma<256, 128><<<dim3(2, 256), 512, 0, stream>>>(xb, eab, eidx, WfT, bf2, WsT, bs2,
                                                          out, E);
}
